// Rwkv7SelfAttention_53223234732576
// MI455X (gfx1250) — matrix operand order repaired and checked by reading, not compiled, not run on hardware
//
#include <hip/hip_runtime.h>
#include <math.h>

// ---------------------------------------------------------------------------
// RWKV7 forward for MI455X (gfx1250). Wave32, WMMA bf16 16x16x32.
// T=2048, H=2048, N=32 heads, D=64.
// ---------------------------------------------------------------------------

typedef unsigned short bfu;                                    // raw bf16 bits
typedef __attribute__((ext_vector_type(16))) __bf16 v16bf;
typedef __attribute__((ext_vector_type(8)))  float  v8f;

union Frag16 { v16bf v; bfu u[16]; uint4 q[2]; };

__device__ __forceinline__ bfu f2bf(float f) {
    union { float f; unsigned u; } c; c.f = f;
    unsigned r = 0x7FFFu + ((c.u >> 16) & 1u);                 // RNE
    return (bfu)((c.u + r) >> 16);
}
__device__ __forceinline__ float sigm(float x) { return 1.0f / (1.0f + __expf(-x)); }

// ---------------------------------------------------------------------------
// Generic elementwise f32 -> bf16 with optional activation.
// mode: 0 = identity, 1 = tanh, 2 = sigmoid
// ---------------------------------------------------------------------------
__global__ __launch_bounds__(256) void act_cast_kernel(const float* __restrict__ src,
                                                       bfu* __restrict__ dst,
                                                       int n, int mode) {
    int i = blockIdx.x * 256 + threadIdx.x;
    if (i >= n) return;
    float v = src[i];
    if (mode == 1) v = tanhf(v);
    else if (mode == 2) v = sigm(v);
    dst[i] = f2bf(v);
}

// ---------------------------------------------------------------------------
// LayerNorm per token (H=2048, 256 threads x 8 elems). Also emits state1_out.
// ---------------------------------------------------------------------------
__global__ __launch_bounds__(256) void ln_kernel(const float* __restrict__ x,
                                                 const float* __restrict__ w,
                                                 const float* __restrict__ b,
                                                 float* __restrict__ xl,
                                                 float* __restrict__ state1_out,
                                                 int T, int H) {
    int t = blockIdx.x;
    const float* xr = x + (size_t)t * H;
    float vals[8], lsum = 0.f, lsq = 0.f;
#pragma unroll
    for (int j = 0; j < 8; ++j) {
        float v = xr[threadIdx.x + j * 256];
        vals[j] = v; lsum += v; lsq += v * v;
    }
    __shared__ float s1[256], s2[256];
    s1[threadIdx.x] = lsum; s2[threadIdx.x] = lsq;
    __syncthreads();
    for (int st = 128; st > 0; st >>= 1) {
        if (threadIdx.x < st) {
            s1[threadIdx.x] += s1[threadIdx.x + st];
            s2[threadIdx.x] += s2[threadIdx.x + st];
        }
        __syncthreads();
    }
    float mu  = s1[0] / (float)H;
    float var = s2[0] / (float)H - mu * mu;
    float inv = rsqrtf(var + 1e-5f);
#pragma unroll
    for (int j = 0; j < 8; ++j) {
        int i = threadIdx.x + j * 256;
        float xn = (vals[j] - mu) * inv * w[i] + b[i];
        xl[(size_t)t * H + i] = xn;
        if (t == T - 1) state1_out[i] = xn + 1.1754943508222875e-38f;
    }
}

// ---------------------------------------------------------------------------
// Token-shift mixing: 6 bf16 activation streams.
// ---------------------------------------------------------------------------
__global__ __launch_bounds__(256) void mix_kernel(const float* __restrict__ xl,
                                                  const float* __restrict__ state1,
                                                  const float* __restrict__ xr_m, const float* __restrict__ xw_m,
                                                  const float* __restrict__ xk_m, const float* __restrict__ xv_m,
                                                  const float* __restrict__ xa_m, const float* __restrict__ xg_m,
                                                  bfu* __restrict__ mr, bfu* __restrict__ mw, bfu* __restrict__ mk,
                                                  bfu* __restrict__ mv, bfu* __restrict__ ma, bfu* __restrict__ mg,
                                                  int H) {
    int t = blockIdx.x;
#pragma unroll
    for (int j = 0; j < 8; ++j) {
        int i = threadIdx.x + j * 256;
        size_t idx = (size_t)t * H + i;
        float cur  = xl[idx];
        float past = (t == 0) ? state1[i] : xl[idx - H];
        float sx = past - cur;
        mr[idx] = f2bf(cur + xr_m[i] * sx);
        mw[idx] = f2bf(cur + xw_m[i] * sx);
        mk[idx] = f2bf(cur + xk_m[i] * sx);
        mv[idx] = f2bf(cur + xv_m[i] * sx);
        ma[idx] = f2bf(cur + xa_m[i] * sx);
        mg[idx] = f2bf(cur + xg_m[i] * sx);
    }
}

// ---------------------------------------------------------------------------
// WMMA bf16 GEMM: C[M,N](f32) = A[M,K](bf16,row) @ B[K,N](bf16,row) (+Cadd)
// Block: 256 threads = 8 waves. Block tile 256(M) x 64(N): each wave owns two
// 16-row A strips (m0, m0+128) and reuses the block's 4 B-fragments for
// 8 WMMAs per 32-deep k-step. Requires M%256==0, K%32==0, N%16==0.
//
// B staged in LDS *transposed* (Bt[col][k]): a WMMA B-fragment is 16
// consecutive K values of one column -> two aligned ds_load_b128 per frag;
// 40-ushort row stride keeps rows 16B-aligned, 20-bank stride = conflict-free.
//
// Manually unrolled x2 with ping-pong A fragments (aP/aQ) and fixed tile
// parities: steady state has no register copies and no WMMA->VALU WAR hazard
// NOPs. One barrier per 32-deep k-step. FULL is a template parameter so the
// hot instantiation has no guard branches in the loop.
// ---------------------------------------------------------------------------
template <bool FULL>
__global__ __launch_bounds__(256) void gemm_bf16_kernel(const bfu* __restrict__ A,
                                                        const bfu* __restrict__ B,
                                                        float* __restrict__ C,
                                                        const float* __restrict__ Cadd,
                                                        int M, int N, int K) {
    __shared__ bfu Bt[2][64][40];
    const int wave = threadIdx.x >> 5;
    const int lane = threadIdx.x & 31;
    const int mrow = lane & 15;
    const int half = lane >> 4;
    const int m0 = blockIdx.x * 256 + wave * 16;     // A strip 0
    const int m1 = m0 + 128;                         // A strip 1
    const int n_base = blockIdx.y * 64;

    const int brow = threadIdx.x >> 3;               // k row 0..31
    const int bc8  = (threadIdx.x & 7) * 8;          // 8-column group base

    auto load_tile = [&](int buf, int k0) {
        if (FULL) {
            union { uint4 q; bfu e[8]; } t;
            t.q = *(const uint4*)(B + (size_t)(k0 + brow) * N + n_base + bc8);
#pragma unroll
            for (int j = 0; j < 8; ++j) Bt[buf][bc8 + j][brow] = t.e[j];
        } else {                                     // only the N=32 LoRA GEMM
#pragma unroll
            for (int j = 0; j < 8; ++j) {
                int col = bc8 + j;
                int gn  = n_base + col;
                Bt[buf][col][brow] = (gn < N) ? B[(size_t)(k0 + brow) * N + gn] : (bfu)0;
            }
        }
    };
    auto load_a = [&](Frag16& f, int m, int k0) {
        const bfu* Ap = A + (size_t)(m + mrow) * K + k0 + half * 16;
        f.q[0] = *(const uint4*)(Ap);
        f.q[1] = *(const uint4*)(Ap + 8);
    };

    v8f acc[8] = {};                                 // acc[nt*2 + strip]

    auto compute = [&](int buf, const Frag16& f0, const Frag16& f1) {
        const int kb = half * 16;
        const bfu* bp0 = &Bt[buf][mrow][kb];
        const bfu* bp1 = &Bt[buf][16 + mrow][kb];
        const bfu* bp2 = &Bt[buf][32 + mrow][kb];
        const bfu* bp3 = &Bt[buf][48 + mrow][kb];
        Frag16 b0, b1, b2, b3;
        b0.q[0] = *(const uint4*)(bp0); b0.q[1] = *(const uint4*)(bp0 + 8);
        b1.q[0] = *(const uint4*)(bp1); b1.q[1] = *(const uint4*)(bp1 + 8);
        b2.q[0] = *(const uint4*)(bp2); b2.q[1] = *(const uint4*)(bp2 + 8);
        b3.q[0] = *(const uint4*)(bp3); b3.q[1] = *(const uint4*)(bp3 + 8);
        acc[0] = __builtin_amdgcn_wmma_f32_16x16x32_bf16(false, f0.v, false, b0.v, (short)0, acc[0], false, false);
        acc[1] = __builtin_amdgcn_wmma_f32_16x16x32_bf16(false, f1.v, false, b0.v, (short)0, acc[1], false, false);
        acc[2] = __builtin_amdgcn_wmma_f32_16x16x32_bf16(false, f0.v, false, b1.v, (short)0, acc[2], false, false);
        acc[3] = __builtin_amdgcn_wmma_f32_16x16x32_bf16(false, f1.v, false, b1.v, (short)0, acc[3], false, false);
        acc[4] = __builtin_amdgcn_wmma_f32_16x16x32_bf16(false, f0.v, false, b2.v, (short)0, acc[4], false, false);
        acc[5] = __builtin_amdgcn_wmma_f32_16x16x32_bf16(false, f1.v, false, b2.v, (short)0, acc[5], false, false);
        acc[6] = __builtin_amdgcn_wmma_f32_16x16x32_bf16(false, f0.v, false, b3.v, (short)0, acc[6], false, false);
        acc[7] = __builtin_amdgcn_wmma_f32_16x16x32_bf16(false, f1.v, false, b3.v, (short)0, acc[7], false, false);
    };

    Frag16 aP0, aP1, aQ0, aQ1;
    load_a(aP0, m0, 0);
    load_a(aP1, m1, 0);
    load_tile(0, 0);
    __syncthreads();

    int k0 = 0;
    for (;;) {
        // ---- even step: consume LDS tile 0 + aP, prefetch tile 1 + aQ ----
        if (k0 + 32 < K) {
            load_tile(1, k0 + 32);
            load_a(aQ0, m0, k0 + 32);
            load_a(aQ1, m1, k0 + 32);
        }
        compute(0, aP0, aP1);
        __syncthreads();
        k0 += 32;
        if (k0 >= K) break;

        // ---- odd step: consume LDS tile 1 + aQ, prefetch tile 0 + aP ----
        if (k0 + 32 < K) {
            load_tile(0, k0 + 32);
            load_a(aP0, m0, k0 + 32);
            load_a(aP1, m1, k0 + 32);
        }
        compute(1, aQ0, aQ1);
        __syncthreads();
        k0 += 32;
        if (k0 >= K) break;
    }

    // store: VGPR vi -> row mbase + half*8 + vi, col ncol + (lane&15)
#pragma unroll
    for (int nt = 0; nt < 4; ++nt) {
        int ncol = n_base + nt * 16;
        if (!FULL && ncol >= N) continue;
#pragma unroll
        for (int strip = 0; strip < 2; ++strip) {
            int mbase = strip ? m1 : m0;
            v8f a = acc[nt * 2 + strip];
#pragma unroll
            for (int vi = 0; vi < 8; ++vi) {
                int mm = mbase + half * 8 + vi;
                size_t idx = (size_t)mm * N + ncol + mrow;
                float val = a[vi];
                if (Cadd) val += Cadd[idx];
                C[idx] = val;
            }
        }
    }
}

// ---------------------------------------------------------------------------
// Per-(token, head) fusion: decay w, in-context lr a_g, kk normalize,
// removal/replacement directions, k & v finalize. 64 threads (one per d).
// In-place: kbuf->k_final, vbuf->v_final, wbuf(pre)->w, abuf(pre)->(-kk),
// bbuf holds vpre on input, b = kk*a_g on output.
// ---------------------------------------------------------------------------
__global__ __launch_bounds__(64) void headprep_kernel(float* __restrict__ kbuf,
                                                      float* __restrict__ vbuf,
                                                      float* __restrict__ wbuf,
                                                      float* __restrict__ abuf,
                                                      float* __restrict__ bbuf,
                                                      const float* __restrict__ v_first,
                                                      const float* __restrict__ w0,
                                                      const float* __restrict__ a0,
                                                      const float* __restrict__ v0,
                                                      const float* __restrict__ k_k,
                                                      const float* __restrict__ k_a,
                                                      int H) {
    int t = blockIdx.x, n = blockIdx.y, d = threadIdx.x;
    int h = n * 64 + d;
    size_t off = (size_t)t * H + h;

    float kg   = kbuf[off];
    float ag   = sigm(a0[h] + abuf[off]);
    float vpre = bbuf[off];
    float kk   = kg * k_k[h];

    __shared__ float sh[64];
    sh[d] = kk * kk;
    __syncthreads();
    float ss = 0.f;
#pragma unroll
    for (int j = 0; j < 64; ++j) ss += sh[j];
    float denom = fmaxf(sqrtf(ss), 1e-6f);
    float kkn = kk / denom;

    float wsg  = sigm(w0[h] + wbuf[off]);
    float wv   = __expf(-0.606531f * wsg);
    float vg   = vbuf[off];
    float vout = vg + (v_first[off] - vg) * sigm(v0[h] + vpre);
    float kf   = kg * (1.0f + (ag - 1.0f) * k_a[h]);

    kbuf[off] = kf;
    vbuf[off] = vout;
    wbuf[off] = wv;
    abuf[off] = -kkn;
    bbuf[off] = kkn * ag;
}

// ---------------------------------------------------------------------------
// Sequential WKV scan. One block per head, 64 threads; lane d owns state row
// S[d][0..63] in VGPRs. Per-step vectors broadcast through LDS.
// ---------------------------------------------------------------------------
__global__ __launch_bounds__(64) void scan_kernel(const float* __restrict__ r,
                                                  const float* __restrict__ k,
                                                  const float* __restrict__ v,
                                                  const float* __restrict__ w,
                                                  const float* __restrict__ adir,
                                                  const float* __restrict__ bdir,
                                                  const float* __restrict__ state2,
                                                  float* __restrict__ y,
                                                  float* __restrict__ s_out,
                                                  int T, int H) {
    int n = blockIdx.x;
    int d = threadIdx.x;
    __shared__ float sh[6][64];

    float S[64];
    const float* S0 = state2 + (size_t)n * 4096 + (size_t)d * 64;
#pragma unroll
    for (int j = 0; j < 64; ++j) S[j] = S0[j];

    for (int t = 0; t < T; ++t) {
        size_t off = (size_t)t * H + n * 64 + d;
        sh[0][d] = r[off];   sh[1][d] = k[off];    sh[2][d] = v[off];
        sh[3][d] = w[off];   sh[4][d] = adir[off]; sh[5][d] = bdir[off];
        __syncthreads();

        float Sa = 0.f;
#pragma unroll
        for (int j = 0; j < 64; ++j) Sa = fmaf(S[j], sh[4][j], Sa);
        float vd = sh[2][d];
        float yv = 0.f;
#pragma unroll
        for (int j = 0; j < 64; ++j) {
            S[j] = fmaf(S[j], sh[3][j], fmaf(Sa, sh[5][j], vd * sh[1][j]));
            yv = fmaf(S[j], sh[0][j], yv);
        }
        y[off] = yv;
        if (t + 8 < T) {
            __builtin_prefetch(&r[off + (size_t)8 * H], 0, 0);   // global_prefetch_b8
            __builtin_prefetch(&w[off + (size_t)8 * H], 0, 0);
        }
        __syncthreads();
    }
    float* So = s_out + (size_t)n * 4096 + (size_t)d * 64;
#pragma unroll
    for (int j = 0; j < 64; ++j) So[j] = S[j];
}

// ---------------------------------------------------------------------------
// Instance-norm over D + bonus + gate -> bf16 o-stream for the final GEMM.
// ---------------------------------------------------------------------------
__global__ __launch_bounds__(64) void outmix_kernel(const float* __restrict__ y,
                                                    const float* __restrict__ r,
                                                    const float* __restrict__ k,
                                                    const float* __restrict__ v,
                                                    const float* __restrict__ g,
                                                    const float* __restrict__ r_k,
                                                    const float* __restrict__ lnw,
                                                    const float* __restrict__ lnb,
                                                    bfu* __restrict__ o, int H) {
    int t = blockIdx.x, n = blockIdx.y, d = threadIdx.x;
    int h = n * 64 + d;
    size_t off = (size_t)t * H + h;
    float yv = y[off];

    __shared__ float s1[64], s2[64], s3[64];
    s1[d] = yv;
    s2[d] = yv * yv;
    s3[d] = r[off] * k[off] * r_k[h];
    __syncthreads();
    float sum = 0.f, sq = 0.f, dot = 0.f;
#pragma unroll
    for (int j = 0; j < 64; ++j) { sum += s1[j]; sq += s2[j]; dot += s3[j]; }
    float mu  = sum * (1.0f / 64.0f);
    float var = sq  * (1.0f / 64.0f) - mu * mu;
    float ov = (yv - mu) * rsqrtf(var + 0.00064f) * lnw[h] + lnb[h];
    float bonus = dot * v[off];
    o[off] = f2bf((ov + bonus) * g[off]);
}

// ---------------------------------------------------------------------------
// Host launcher
// ---------------------------------------------------------------------------
extern "C" void kernel_launch(void* const* d_in, const int* in_sizes, int n_in,
                              void* d_out, int out_size, void* d_ws, size_t ws_size,
                              hipStream_t stream) {
    (void)in_sizes; (void)n_in; (void)out_size; (void)ws_size;
    const int T = 2048, H = 2048, N = 32;
    const int DW = 64, DA = 64, DV = 32, DG = 128;
    const size_t TH = (size_t)T * H;

    const float* x      = (const float*)d_in[0];
    const float* state1 = (const float*)d_in[1];
    const float* state2 = (const float*)d_in[2];
    const float* vfirst = (const float*)d_in[3];
    const float* ln1w   = (const float*)d_in[4];
    const float* ln1b   = (const float*)d_in[5];
    const float* x_r    = (const float*)d_in[6];
    const float* x_w    = (const float*)d_in[7];
    const float* x_k    = (const float*)d_in[8];
    const float* x_v    = (const float*)d_in[9];
    const float* x_a    = (const float*)d_in[10];
    const float* x_g    = (const float*)d_in[11];
    const float* w0     = (const float*)d_in[12];
    const float* w1     = (const float*)d_in[13];
    const float* w2     = (const float*)d_in[14];
    const float* a0     = (const float*)d_in[15];
    const float* a1     = (const float*)d_in[16];
    const float* a2     = (const float*)d_in[17];
    const float* v0     = (const float*)d_in[18];
    const float* v1     = (const float*)d_in[19];
    const float* v2     = (const float*)d_in[20];
    const float* g1     = (const float*)d_in[21];
    const float* g2     = (const float*)d_in[22];
    const float* Wr     = (const float*)d_in[23];
    const float* Wk     = (const float*)d_in[24];
    const float* Wv     = (const float*)d_in[25];
    const float* Wo     = (const float*)d_in[26];
    const float* k_k    = (const float*)d_in[27];
    const float* k_a    = (const float*)d_in[28];
    const float* r_k    = (const float*)d_in[29];
    const float* ln_x_w = (const float*)d_in[30];
    const float* ln_x_b = (const float*)d_in[31];

    float* out        = (float*)d_out;
    float* state1_out = out + TH;
    float* s_out      = out + TH + H;
    float* vfirst_out = out + TH + H + (size_t)N * 64 * 64;

    // ---- workspace bump allocator (256B aligned) ----
    char* ws = (char*)d_ws;
    size_t off = 0;
    auto alloc = [&](size_t bytes) -> char* {
        off = (off + 255) & ~(size_t)255;
        char* p = ws + off;
        off += bytes;
        return p;
    };
    float* xl   = (float*)alloc(TH * 4);
    bfu* mr = (bfu*)alloc(TH * 2); bfu* mw = (bfu*)alloc(TH * 2);
    bfu* mk = (bfu*)alloc(TH * 2); bfu* mv = (bfu*)alloc(TH * 2);
    bfu* ma = (bfu*)alloc(TH * 2); bfu* mg = (bfu*)alloc(TH * 2);
    bfu* Wr_b = (bfu*)alloc((size_t)H * H * 2);
    bfu* Wk_b = (bfu*)alloc((size_t)H * H * 2);
    bfu* Wv_b = (bfu*)alloc((size_t)H * H * 2);
    bfu* Wo_b = (bfu*)alloc((size_t)H * H * 2);
    bfu* w1_b = (bfu*)alloc((size_t)H * DW * 2); bfu* w2_b = (bfu*)alloc((size_t)DW * H * 2);
    bfu* a1_b = (bfu*)alloc((size_t)H * DA * 2); bfu* a2_b = (bfu*)alloc((size_t)DA * H * 2);
    bfu* v1_b = (bfu*)alloc((size_t)H * DV * 2); bfu* v2_b = (bfu*)alloc((size_t)DV * H * 2);
    bfu* g1_b = (bfu*)alloc((size_t)H * DG * 2); bfu* g2_b = (bfu*)alloc((size_t)DG * H * 2);
    float* rbuf = (float*)alloc(TH * 4);
    float* kbuf = (float*)alloc(TH * 4);
    float* vbuf = (float*)alloc(TH * 4);
    float* hw   = (float*)alloc((size_t)T * DW * 4);
    float* ha   = (float*)alloc((size_t)T * DA * 4);
    float* hv   = (float*)alloc((size_t)T * DV * 4);
    float* hg   = (float*)alloc((size_t)T * DG * 4);
    bfu* thw_b  = (bfu*)alloc((size_t)T * DW * 2);
    bfu* ha_b   = (bfu*)alloc((size_t)T * DA * 2);
    bfu* hv_b   = (bfu*)alloc((size_t)T * DV * 2);
    bfu* shg_b  = (bfu*)alloc((size_t)T * DG * 2);
    float* wbuf = (float*)alloc(TH * 4);   // w-pre, then w
    float* abuf = (float*)alloc(TH * 4);   // a-pre, then -kk
    float* bbuf = (float*)alloc(TH * 4);   // v-pre, then b = kk*a_g
    float* gbuf = (float*)alloc(TH * 4);
    float* ybuf = (float*)alloc(TH * 4);
    bfu* obuf   = (bfu*)alloc(TH * 2);

    auto cast = [&](const float* src, bfu* dst, size_t n, int mode) {
        act_cast_kernel<<<dim3((unsigned)((n + 255) / 256)), 256, 0, stream>>>(src, dst, (int)n, mode);
    };
    auto gemm = [&](const bfu* A, const bfu* B, float* C, const float* Cadd, int M, int Nn, int K) {
        dim3 grid(M / 256, (Nn + 63) / 64);
        if ((Nn & 63) == 0)
            gemm_bf16_kernel<true><<<grid, 256, 0, stream>>>(A, B, C, Cadd, M, Nn, K);
        else
            gemm_bf16_kernel<false><<<grid, 256, 0, stream>>>(A, B, C, Cadd, M, Nn, K);
    };

    // 1. weights -> bf16
    cast(Wr, Wr_b, (size_t)H * H, 0); cast(Wk, Wk_b, (size_t)H * H, 0);
    cast(Wv, Wv_b, (size_t)H * H, 0); cast(Wo, Wo_b, (size_t)H * H, 0);
    cast(w1, w1_b, (size_t)H * DW, 0); cast(w2, w2_b, (size_t)DW * H, 0);
    cast(a1, a1_b, (size_t)H * DA, 0); cast(a2, a2_b, (size_t)DA * H, 0);
    cast(v1, v1_b, (size_t)H * DV, 0); cast(v2, v2_b, (size_t)DV * H, 0);
    cast(g1, g1_b, (size_t)H * DG, 0); cast(g2, g2_b, (size_t)DG * H, 0);

    // 2. LN + state1_out, 3. token-shift mix
    ln_kernel<<<T, 256, 0, stream>>>(x, ln1w, ln1b, xl, state1_out, T, H);
    mix_kernel<<<T, 256, 0, stream>>>(xl, state1, x_r, x_w, x_k, x_v, x_a, x_g,
                                      mr, mw, mk, mv, ma, mg, H);

    // 4. big projections + LoRA stage 1 (all WMMA)
    gemm(mr, Wr_b, rbuf, nullptr, T, H, H);
    gemm(mk, Wk_b, kbuf, nullptr, T, H, H);
    gemm(mv, Wv_b, vbuf, nullptr, T, H, H);
    gemm(mw, w1_b, hw, nullptr, T, DW, H);
    gemm(ma, a1_b, ha, nullptr, T, DA, H);
    gemm(mv, v1_b, hv, nullptr, T, DV, H);
    gemm(mg, g1_b, hg, nullptr, T, DG, H);

    // 5. inner activations -> bf16
    cast(hw, thw_b, (size_t)T * DW, 1);   // tanh
    cast(ha, ha_b,  (size_t)T * DA, 0);   // identity
    cast(hv, hv_b,  (size_t)T * DV, 0);   // identity
    cast(hg, shg_b, (size_t)T * DG, 2);   // sigmoid

    // 6. LoRA stage 2 (WMMA)
    gemm(thw_b, w2_b, wbuf, nullptr, T, H, DW);
    gemm(ha_b,  a2_b, abuf, nullptr, T, H, DA);
    gemm(hv_b,  v2_b, bbuf, nullptr, T, H, DV);
    gemm(shg_b, g2_b, gbuf, nullptr, T, H, DG);

    // 7. per-(token,head) fuse
    headprep_kernel<<<dim3(T, N), 64, 0, stream>>>(kbuf, vbuf, wbuf, abuf, bbuf,
                                                   vfirst, w0, a0, v0, k_k, k_a, H);

    // 8. sequential WKV scan (writes S_out)
    scan_kernel<<<N, 64, 0, stream>>>(rbuf, kbuf, vbuf, wbuf, abuf, bbuf,
                                      state2, ybuf, s_out, T, H);

    // 9. instance-norm + bonus + gate -> bf16
    outmix_kernel<<<dim3(T, N), 64, 0, stream>>>(ybuf, rbuf, kbuf, vbuf, gbuf,
                                                 r_k, ln_x_w, ln_x_b, obuf, H);

    // 10. output projection fused with residual add (WMMA), straight into d_out
    gemm(obuf, Wo_b, out, x, T, H, H);

    // 11. v_first passthrough
    hipMemcpyAsync(vfirst_out, (const void*)vfirst, TH * sizeof(float),
                   hipMemcpyDeviceToDevice, stream);
}
